// EdgeHead_44023414784043
// MI455X (gfx1250) — matrix operand order repaired and checked by reading, not compiled, not run on hardware
//
#include <hip/hip_runtime.h>

typedef __attribute__((ext_vector_type(16))) _Float16 v16h;
typedef __attribute__((ext_vector_type(8)))  _Float16 v8h;
typedef __attribute__((ext_vector_type(4)))  _Float16 v4h;
typedef __attribute__((ext_vector_type(8)))  float    v8f;

#define WAVES  4
#define TPB    (WAVES * 32)
#define ROWPAD 264            // 256 + 8 halfs of padding (528B rows -> bank-rotated)
#define LN_EPS 1e-5f

union V16U { v16h v; v8h h[2]; };

__global__ __launch_bounds__(TPB)
void edgehead_wmma_kernel(const float* __restrict__ h,
                          const int*   __restrict__ ep,      // [2, E]
                          const float* __restrict__ gamma,
                          const float* __restrict__ beta,
                          const float* __restrict__ W1,      // [256, 32] row-major
                          const float* __restrict__ b1,      // [32]
                          const float* __restrict__ W2,      // [32]
                          const float* __restrict__ b2,      // [1]
                          float* __restrict__ out,           // [E]
                          int E)
{
    // Block-shared, loaded once:
    __shared__ __align__(16) _Float16 w1h[8192];   // W1 as f16, WMMA-B fragment order
    __shared__ __align__(16) float gammaB[256];
    __shared__ __align__(16) float betaB[256];
    __shared__ __align__(16) float b1B[32];
    __shared__ __align__(16) float w2B[32];
    __shared__ float b2B;
    // Per-wave scratch:
    __shared__ __align__(16) _Float16 featA[WAVES][16 * ROWPAD]; // f16 feature rows
    __shared__ __align__(16) float    hmid [WAVES][16 * 33];     // relu(hidden)

    const int tid = threadIdx.x;

    // ---- Preload W1 into B-fragment layout ----
    // w1h index = ((t*8 + kk)*32 + L)*16 + i  holds W1[k][n] with
    //   k = kk*32 + (L>=16 ? 16 : 0) + i,  n = t*16 + (L & 15)
    for (int idx = tid; idx < 8192; idx += TPB) {
        int i  = idx & 15;
        int L  = (idx >> 4) & 31;
        int kk = (idx >> 9) & 7;
        int t  = idx >> 12;
        int k  = kk * 32 + ((L >> 4) << 4) + i;
        int n  = t * 16 + (L & 15);
        w1h[idx] = (_Float16)W1[k * 32 + n];
    }
    for (int idx = tid; idx < 256; idx += TPB) {
        gammaB[idx] = gamma[idx];
        betaB[idx]  = beta[idx];
    }
    if (tid < 32) { b1B[tid] = b1[tid]; w2B[tid] = W2[tid]; }
    if (tid == 0) b2B = b2[0];
    __syncthreads();

    const int lane = tid & 31;
    const int wv   = tid >> 5;
    const int r    = lane & 15;   // edge slot in tile / matrix row M
    const int hh   = lane >> 4;   // which 32-feature half this lane gathers
    const int numTiles   = (E + 15) >> 4;
    const int waveGlobal = blockIdx.x * WAVES + wv;
    const int waveStride = gridDim.x * WAVES;

    _Float16* myrow = &featA[wv][r * ROWPAD];
    float*    hmw   = hmid[wv];

    for (int tile = waveGlobal; tile < numTiles; tile += waveStride) {
        const int e0 = tile << 4;
        int e = e0 + r; if (e >= E) e = E - 1;           // clamp, store is guarded
        const int src = ep[e];
        const int dst = ep[E + e];
        const float* pu = h + (size_t)src * 64 + hh * 32;
        const float* pv = h + (size_t)dst * 64 + hh * 32;

        float4 ua[8], va[8];
        #pragma unroll
        for (int g = 0; g < 8; ++g) {
            ua[g] = ((const float4*)pu)[g];
            va[g] = ((const float4*)pv)[g];
        }

        // ---- LayerNorm statistics over this lane's 128 feature values ----
        float s1 = 0.f, s2 = 0.f;
        #pragma unroll
        for (int g = 0; g < 8; ++g) {
            float a4[4] = {ua[g].x, ua[g].y, ua[g].z, ua[g].w};
            float b4[4] = {va[g].x, va[g].y, va[g].z, va[g].w};
            #pragma unroll
            for (int q = 0; q < 4; ++q) {
                float dd = fabsf(a4[q] - b4[q]);
                float pp = a4[q] * b4[q];
                s1 += a4[q] + b4[q] + dd + pp;
                s2 += a4[q]*a4[q] + b4[q]*b4[q] + dd*dd + pp*pp;
            }
        }
        s1 += __shfl_xor(s1, 16, 32);   // combine the edge's two lanes
        s2 += __shfl_xor(s2, 16, 32);
        const float mu   = s1 * (1.0f / 256.0f);
        const float var  = s2 * (1.0f / 256.0f) - mu * mu;
        const float rstd = rsqrtf(var + LN_EPS);

        // ---- Normalize, scale/shift, convert to f16, write feature row ----
        #pragma unroll
        for (int s = 0; s < 4; ++s) {
            #pragma unroll
            for (int g = 0; g < 8; ++g) {
                float a4[4] = {ua[g].x, ua[g].y, ua[g].z, ua[g].w};
                float b4[4] = {va[g].x, va[g].y, va[g].z, va[g].w};
                float x[4];
                #pragma unroll
                for (int q = 0; q < 4; ++q) {
                    float xa = a4[q], xb = b4[q];
                    x[q] = (s == 0) ? xa
                         : (s == 1) ? xb
                         : (s == 2) ? fabsf(xa - xb)
                                    : xa * xb;
                }
                const int k = s * 64 + hh * 32 + g * 4;
                float4 gm = *(const float4*)&gammaB[k];
                float4 bt = *(const float4*)&betaB[k];
                v4h y;
                y[0] = (_Float16)((x[0] - mu) * rstd * gm.x + bt.x);
                y[1] = (_Float16)((x[1] - mu) * rstd * gm.y + bt.y);
                y[2] = (_Float16)((x[2] - mu) * rstd * gm.z + bt.z);
                y[3] = (_Float16)((x[3] - mu) * rstd * gm.w + bt.w);
                *(v4h*)(myrow + k) = y;
            }
        }
        asm volatile("s_wait_dscnt 0" ::: "memory");

        // ---- feat[16x256] @ W1[256x32] via 16x v_wmma_f32_16x16x32_f16 ----
        const float bias0 = b1B[r], bias1 = b1B[16 + r];
        v8f acc0, acc1;
        #pragma unroll
        for (int j = 0; j < 8; ++j) { acc0[j] = bias0; acc1[j] = bias1; }

        #pragma unroll
        for (int kk = 0; kk < 8; ++kk) {
            // A fragment: row M = r, contiguous run [kbase+16*hh, +16)
            const int c0 = kk * 32 + (hh << 4);
            V16U a;
            a.h[0] = *(const v8h*)(myrow + c0);
            a.h[1] = *(const v8h*)(myrow + c0 + 8);
            // B fragments (prearranged, lane-contiguous 32B chunks)
            const _Float16* bp0 = &w1h[(kk * 32 + lane) * 16];
            const _Float16* bp1 = &w1h[((8 + kk) * 32 + lane) * 16];
            V16U bA, bB;
            bA.h[0] = *(const v8h*)bp0; bA.h[1] = *(const v8h*)(bp0 + 8);
            bB.h[0] = *(const v8h*)bp1; bB.h[1] = *(const v8h*)(bp1 + 8);

            acc0 = __builtin_amdgcn_wmma_f32_16x16x32_f16(
                false, a.v, false, bA.v, (short)0, acc0, false, false);
            acc1 = __builtin_amdgcn_wmma_f32_16x16x32_f16(
                false, a.v, false, bB.v, (short)0, acc1, false, false);
        }

        // ---- ReLU + stash hidden [16 x 32]: lane holds N=r(+16t), M = hh*8 + j ----
        const int mbase = hh * 8;
        #pragma unroll
        for (int j = 0; j < 8; ++j) {
            float x0 = acc0[j]; x0 = x0 > 0.f ? x0 : 0.f;
            float x1 = acc1[j]; x1 = x1 > 0.f ? x1 : 0.f;
            hmw[(mbase + j) * 33 + r]      = x0;
            hmw[(mbase + j) * 33 + 16 + r] = x1;
        }
        asm volatile("s_wait_dscnt 0" ::: "memory");

        // ---- Final [32]->1 projection, one edge per lane 0..15 ----
        if (hh == 0) {
            float acc = b2B;
            #pragma unroll
            for (int n = 0; n < 32; ++n) acc += hmw[r * 33 + n] * w2B[n];
            if (e0 + r < E) out[e0 + r] = acc;
        }
        asm volatile("s_wait_dscnt 0" ::: "memory");  // WAR guard before tile reuse
    }
}

extern "C" void kernel_launch(void* const* d_in, const int* in_sizes, int n_in,
                              void* d_out, int out_size, void* d_ws, size_t ws_size,
                              hipStream_t stream) {
    const float* h     = (const float*)d_in[0];
    const int*   ep    = (const int*)  d_in[1];
    const float* gamma = (const float*)d_in[2];
    const float* beta  = (const float*)d_in[3];
    const float* W1    = (const float*)d_in[4];
    const float* b1    = (const float*)d_in[5];
    const float* W2    = (const float*)d_in[6];
    const float* b2    = (const float*)d_in[7];
    float* out = (float*)d_out;
    const int E = out_size;

    int numTiles = (E + 15) >> 4;
    int blocks = (numTiles + WAVES - 1) / WAVES;
    if (blocks > 2048) blocks = 2048;
    if (blocks < 1)    blocks = 1;

    edgehead_wmma_kernel<<<blocks, TPB, 0, stream>>>(
        h, ep, gamma, beta, W1, b1, W2, b2, out, E);
}
